// GATRouterActor_627065225236
// MI455X (gfx1250) — matrix operand order repaired and checked by reading, not compiled, not run on hardware
//
#include <hip/hip_runtime.h>

typedef unsigned short u16;
typedef unsigned char  u8;
typedef __attribute__((ext_vector_type(16))) __bf16 v16bf;
typedef __attribute__((ext_vector_type(8)))  float  v8f;

#define NB   8      // batch
#define NV   1024   // vehicles
#define NR   512    // routes
#define NU   16
#define NH   4      // heads
#define HID  128
#define OUTD 32
#define ST_STRIDE 7768   // (V+R+U)*5 + GLOB

// ---------- bf16 helpers (round-to-nearest-even) ----------
__device__ __forceinline__ u16 f2bf(float f) {
  union { float f; unsigned u; } x; x.f = f;
  unsigned r = x.u + 0x7FFFu + ((x.u >> 16) & 1u);
  return (u16)(r >> 16);
}
__device__ __forceinline__ float bf2f(u16 b) {
  union { unsigned u; float f; } x; x.u = ((unsigned)b) << 16;
  return x.f;
}

// ============================================================
// Generic batched bf16 GEMM:  C[z] = A[z][M,K] * Bt[z][N,K]^T (+bias, relu)
//   A  : row-major [M,K]   (ldA)
//   Bt : row-major [N,K]   (ldB)  -- B transposed; per-lane fragment is
//                                    two runs of 8 contiguous K values (kh.., kh+16..) -> two b128 loads.
// wave = 32x32 output tile (4 wmma accumulators), block = 4 waves = 128 rows.
// Out-of-range A rows are CLAMPED to row M-1 (they only feed output rows the
// store masks off), so the K-loop is branch-free: 8 b128 loads + 4 WMMA.
// Per-batch operand offset = (z/innerN)*Outer + (z%innerN)*Inner (elements).
// Requires K%32==0, N%32==0; M arbitrary.
// cbT!=0 : store bf16 C transposed: Cb[col*ldC + row]  (for Whs producers).
// ============================================================
__global__ void __launch_bounds__(128)
gemm_bf16_wmma(const u16* __restrict__ A, const u16* __restrict__ Bt,
               const float* __restrict__ bias,
               float* __restrict__ Cf, u16* __restrict__ Cb,
               int M, int N, int K, int ldA, int ldB, int ldC,
               int innerN,
               long aO, long aI, long bO, long bI, long cO, long cI,
               int relu, int cbT)
{
  const int  z  = blockIdx.z;
  const long zo = z / innerN, zi = z % innerN;
  const u16* Ab = A  + zo * aO + zi * aI;
  const u16* Bb = Bt + zo * bO + zi * bI;

  const int lane = threadIdx.x & 31;
  const int wave = threadIdx.x >> 5;
  const int m0 = blockIdx.x * 128 + wave * 32;
  const int n0 = blockIdx.y * 32;

  // A fragment (16-bit A 16x32, ISA 7.12.2): lane&15 = row, K-half = 8*(lane>=16);
  // per lane ks = {kh..kh+7, kh+16..kh+23} -> two contiguous 16B chunks.
  const int arow0 = m0 + (lane & 15);
  const int arow1 = arow0 + 16;
  const int ar0c  = (arow0 < M) ? arow0 : (M - 1);   // clamp, see header comment
  const int ar1c  = (arow1 < M) ? arow1 : (M - 1);
  const int akh   = (lane >> 4) * 8;
  // B fragment (16-bit B 32x16): lane&15 = col, ks = {kh..kh+7, kh+16..kh+23}, kh = 8*(lane>=16)
  // -> two contiguous 16B chunks per lane (same k order as A).
  const int bn  = n0 + (lane & 15);
  const int bkh = (lane >> 4) * 8;

  const u16* ap0 = Ab + (long)ar0c * ldA + akh;
  const u16* ap1 = Ab + (long)ar1c * ldA + akh;
  const u16* bp0 = Bb + (long)bn * ldB + bkh;
  const u16* bp1 = bp0 + (long)16 * ldB;

  v8f acc00 = {0.f,0.f,0.f,0.f,0.f,0.f,0.f,0.f};
  v8f acc01 = acc00, acc10 = acc00, acc11 = acc00;

  for (int kk = 0; kk < K; kk += 32) {
    union Frag { v16bf v; uint4 q[2]; u16 s[16]; } a0, a1, b0, b1;
    a0.q[0] = *(const uint4*)(ap0 + kk);      a0.q[1] = *(const uint4*)(ap0 + kk + 16);
    a1.q[0] = *(const uint4*)(ap1 + kk);      a1.q[1] = *(const uint4*)(ap1 + kk + 16);
    b0.q[0] = *(const uint4*)(bp0 + kk);      b0.q[1] = *(const uint4*)(bp0 + kk + 16);
    b1.q[0] = *(const uint4*)(bp1 + kk);      b1.q[1] = *(const uint4*)(bp1 + kk + 16);

    __builtin_prefetch(ap0 + kk + 32, 0, 3);  // speculative; dropped if invalid

    acc00 = __builtin_amdgcn_wmma_f32_16x16x32_bf16(false, a0.v, false, b0.v,
                                                    (short)0, acc00, false, false);
    acc01 = __builtin_amdgcn_wmma_f32_16x16x32_bf16(false, a0.v, false, b1.v,
                                                    (short)0, acc01, false, false);
    acc10 = __builtin_amdgcn_wmma_f32_16x16x32_bf16(false, a1.v, false, b0.v,
                                                    (short)0, acc10, false, false);
    acc11 = __builtin_amdgcn_wmma_f32_16x16x32_bf16(false, a1.v, false, b1.v,
                                                    (short)0, acc11, false, false);
  }

  // C layout (32-bit 16x16): lane&15 = col, vgpr r -> row r (+8 for lanes>=16)
  float* Cfb = Cf ? (Cf + zo * cO + zi * cI) : nullptr;
  u16*   Cbb = Cb ? (Cb + zo * cO + zi * cI) : nullptr;
  const int ccol  = n0 + (lane & 15);
  const int rbase = m0 + ((lane >> 4) ? 8 : 0);
  const float bia0 = bias ? bias[ccol]      : 0.f;
  const float bia1 = bias ? bias[ccol + 16] : 0.f;
  const bool full = (m0 + 32 <= M);
#pragma unroll
  for (int half = 0; half < 2; ++half) {
    const v8f* c0 = half ? &acc10 : &acc00;
    const v8f* c1 = half ? &acc11 : &acc01;
#pragma unroll
    for (int r = 0; r < 8; ++r) {
      int row = rbase + half * 16 + r;
      if (full || row < M) {
        float v0 = (*c0)[r] + bia0;
        float v1 = (*c1)[r] + bia1;
        if (relu) { v0 = fmaxf(v0, 0.f); v1 = fmaxf(v1, 0.f); }
        if (cbT) {          // transposed bf16 store: [N, Mtotal]
          Cbb[(long)ccol * ldC + row]        = f2bf(v0);
          Cbb[(long)(ccol + 16) * ldC + row] = f2bf(v1);
        } else {
          long i0 = (long)row * ldC + ccol;
          if (Cfb) { Cfb[i0] = v0;       Cfb[i0 + 16] = v1; }
          if (Cbb) { Cbb[i0] = f2bf(v0); Cbb[i0 + 16] = f2bf(v1); }
        }
      }
    }
  }
}

// ============================================================
// Small-K (<=8) linear + ReLU -> bf16. rows laid out [B, NN] with batch stride.
// ============================================================
__global__ void proj_small(const float* __restrict__ xbase, int Kin, int NN, int ldBatch,
                           const float* __restrict__ W, const float* __restrict__ bias,
                           u16* __restrict__ out, int ldOut, int total)
{
  int idx = blockIdx.x * 256 + threadIdx.x;
  if (idx >= total) return;
  int j = idx & 127, n = idx >> 7;
  int b = n / NN, r = n - b * NN;
  const float* x = xbase + (long)b * ldBatch + (long)r * Kin;
  float acc = bias[j];
  for (int k = 0; k < Kin; ++k) acc += x[k] * W[j * Kin + k];
  out[(long)n * ldOut + j] = f2bf(fmaxf(acc, 0.f));
}

// ============================================================
// GAT prep: fold attention vectors into weights.
// wfs[h,i] = sum_o W[h,i,o]*a[h,o] ; wft uses a[h,32+o]
// c[h,f]   = sum_e a[h,64+e]*We[e,f] ; d[h] = sum_e a[h,64+e]*be[e]
// ============================================================
__global__ void gat_prep(const float* __restrict__ W, const float* __restrict__ a,
                         const float* __restrict__ eW, const float* __restrict__ eb,
                         float* __restrict__ wfs, float* __restrict__ wft,
                         float* __restrict__ cf, float* __restrict__ df)
{
  int tid = threadIdx.x;
  if (tid < 512) {
    int h = tid >> 7, i = tid & 127;
    float ss = 0.f, st = 0.f;
    for (int o = 0; o < OUTD; ++o) {
      float w = W[((long)h * HID + i) * OUTD + o];
      ss += w * a[h * 72 + o];
      st += w * a[h * 72 + 32 + o];
    }
    wfs[tid] = ss; wft[tid] = st;
  }
  if (tid < 32) {
    int h = tid >> 3, f = tid & 7;
    float c = 0.f;
    for (int e = 0; e < 8; ++e) c += a[h * 72 + 64 + e] * eW[e * 8 + f];
    cf[tid] = c;
  }
  if (tid < 4) {
    float dd = 0.f;
    for (int e = 0; e < 8; ++e) dd += a[tid * 72 + 64 + e] * eb[e];
    df[tid] = dd;
  }
}

// pack GAT W [H,HID,OUT] -> bf16 Bt [N=H*OUT, K=HID]
__global__ void pack_gatW(const float* __restrict__ W, u16* __restrict__ out)
{
  int idx = blockIdx.x * 256 + threadIdx.x;
  if (idx >= HID * HID) return;
  int n = idx >> 7, i = idx & 127, h = n >> 5, o = n & 31;
  out[idx] = f2bf(W[((long)h * HID + i) * OUTD + o]);
}

// plain fp32 -> bf16 cast (linear weights are already [Nout, Kin] = Bt layout)
__global__ void pack_cast(const float* __restrict__ in, u16* __restrict__ out, int total)
{
  int idx = blockIdx.x * 256 + threadIdx.x;
  if (idx >= total) return;
  out[idx] = f2bf(in[idx]);
}

// scores: out[(b*H+h)*L + s] = dot(hbf[row s], wfold[h])
__global__ void fold_score(const u16* __restrict__ hbf, const float* __restrict__ wfold,
                           float* __restrict__ out, int L, int total)
{
  int idx = blockIdx.x * 256 + threadIdx.x;
  if (idx >= total) return;
  int h = idx & 3, bs = idx >> 2;
  const u16* row = hbf + (long)bs * HID;
  const float* wf = wfold + h * HID;
  float acc = 0.f;
  for (int i = 0; i < HID; ++i) acc += bf2f(row[i]) * wf[i];
  int b = bs / L, s = bs - b * L;
  out[((long)b * NH + h) * L + s] = acc;
}

// ============================================================
// Edge score + transpose: reads edge[b,s,t,0..7] coalesced over t,
// writes e[b,h,t,s] (bf16, LeakyReLU'd, masked) coalesced over s.
// Block 256 threads handles a 32t x 32s tile.
// ============================================================
__global__ void __launch_bounds__(256)
edge_score(const float* __restrict__ edge, const u8* __restrict__ mask,
           const float* __restrict__ cf, const float* __restrict__ df,
           const float* __restrict__ es, const float* __restrict__ et,
           u16* __restrict__ ebuf, int S, int T)
{
  __shared__ float tile[NH][32][33];
  const int b  = blockIdx.z;
  const int t0 = blockIdx.x * 32;
  const int s0 = blockIdx.y * 32;
  const int tl  = threadIdx.x & 31;
  const int sl0 = threadIdx.x >> 5;
#pragma unroll
  for (int p = 0; p < 4; ++p) {
    int s = s0 + sl0 + p * 8;
    long base = (((long)b * S + s) * T + (t0 + tl)) * 8;
    float4 e0 = *(const float4*)(edge + base);
    float4 e1 = *(const float4*)(edge + base + 4);
    bool mk = mask[((long)b * S + s) * T + (t0 + tl)] != 0;
#pragma unroll
    for (int h = 0; h < NH; ++h) {
      const float* c = cf + h * 8;
      float v = e0.x*c[0] + e0.y*c[1] + e0.z*c[2] + e0.w*c[3]
              + e1.x*c[4] + e1.y*c[5] + e1.z*c[6] + e1.w*c[7] + df[h];
      tile[h][sl0 + p * 8][tl] = mk ? v : -1e30f;
    }
  }
  __syncthreads();
#pragma unroll
  for (int i = 0; i < 16; ++i) {
    int idx = i * 256 + threadIdx.x;
    int sl = idx & 31, tl2 = (idx >> 5) & 31, h = idx >> 10;
    float v = tile[h][sl][tl2]
            + es[((long)b * NH + h) * S + s0 + sl]
            + et[((long)b * NH + h) * T + t0 + tl2];
    v = (v > 0.f) ? v : 0.2f * v;
    ebuf[(((long)b * NH + h) * T + (t0 + tl2)) * S + s0 + sl] = f2bf(v);
  }
}

// ============================================================
// Row softmax over S (<=512): one wave per (b,h,t) row.
// mode 0: in-place on prebuilt e rows; mode 1: e = lrelu(es[s]+et[t]), no mask.
// ============================================================
__global__ void __launch_bounds__(256)
softmax_rows(u16* __restrict__ att, const float* __restrict__ es,
             const float* __restrict__ et, int S, int T, int mode, int rows)
{
  int row = blockIdx.x * 8 + (threadIdx.x >> 5);
  if (row >= rows) return;
  int lane = threadIdx.x & 31;
  int nch = S >> 5;               // S/32 <= 16
  float vals[16];
  float m = -3.4e38f;
  if (mode == 0) {
    const u16* p = att + (long)row * S;
    for (int i = 0; i < nch; ++i) { vals[i] = bf2f(p[lane + i * 32]); m = fmaxf(m, vals[i]); }
  } else {
    int bh = row / T, t = row - bh * T;
    float e_t = et[(long)bh * T + t];
    for (int i = 0; i < nch; ++i) {
      float v = es[(long)bh * S + lane + i * 32] + e_t;
      v = (v > 0.f) ? v : 0.2f * v;
      vals[i] = v; m = fmaxf(m, v);
    }
  }
  for (int off = 16; off > 0; off >>= 1) m = fmaxf(m, __shfl_xor(m, off, 32));
  float sum = 0.f;
  for (int i = 0; i < nch; ++i) { vals[i] = __expf(vals[i] - m); sum += vals[i]; }
  for (int off = 16; off > 0; off >>= 1) sum += __shfl_xor(sum, off, 32);
  float inv = 1.0f / sum;
  u16* p = att + (long)row * S;
  for (int i = 0; i < nch; ++i) p[lane + i * 32] = f2bf(vals[i] * inv);
}

// mean over axis-1 of [B,N,128] fp32 -> bf16 slice of z [B,512]
__global__ void mean_rows(const float* __restrict__ in, u16* __restrict__ z,
                          int seg, int N)
{
  int idx = blockIdx.x * 256 + threadIdx.x;
  if (idx >= NB * HID) return;
  int b = idx >> 7, j = idx & 127;
  float s = 0.f;
  for (int n = 0; n < N; ++n) s += in[((long)b * N + n) * HID + j];
  z[b * 512 + seg + j] = f2bf(s / (float)N);
}

// ============================================================
extern "C" void kernel_launch(void* const* d_in, const int* in_sizes, int n_in,
                              void* d_out, int out_size, void* d_ws, size_t ws_size,
                              hipStream_t stream) {
  (void)in_sizes; (void)n_in; (void)out_size; (void)ws_size;
  const float* state    = (const float*)d_in[0];
  const u8*    vr_mask  = (const u8*)d_in[1];
  const u8*    rr_mask  = (const u8*)d_in[2];
  const float* vr_edge  = (const float*)d_in[3];
  const float* rr_edge  = (const float*)d_in[4];
  // params (depth-first leaf order)
  const float* vpW = (const float*)d_in[5],  *vpB = (const float*)d_in[6];
  const float* rpW = (const float*)d_in[7],  *rpB = (const float*)d_in[8];
  const float* gaW = (const float*)d_in[9],  *gaA = (const float*)d_in[10];
  const float* gaeW= (const float*)d_in[11], *gaeB= (const float*)d_in[12];
  const float* voW = (const float*)d_in[13], *voB = (const float*)d_in[14];
  const float* r2W = (const float*)d_in[15], *r2B = (const float*)d_in[16];
  const float* gpW = (const float*)d_in[17], *gpA = (const float*)d_in[18];
  const float* gpeW= (const float*)d_in[19], *gpeB= (const float*)d_in[20];
  const float* gcW = (const float*)d_in[21], *gcA = (const float*)d_in[22];
  const float* gceW= (const float*)d_in[23], *gceB= (const float*)d_in[24];
  const float* f1W = (const float*)d_in[25], *f1B = (const float*)d_in[26];
  const float* f2W = (const float*)d_in[27], *f2B = (const float*)d_in[28];
  const float* u1W = (const float*)d_in[29], *u1B = (const float*)d_in[30];
  const float* u2W = (const float*)d_in[31], *u2B = (const float*)d_in[32];
  const float* g1W = (const float*)d_in[33], *g1B = (const float*)d_in[34];
  const float* fn1W= (const float*)d_in[35], *fn1B= (const float*)d_in[36];
  const float* fn2W= (const float*)d_in[37], *fn2B= (const float*)d_in[38];

  // ---- workspace carve-up ----
  u8* w = (u8*)d_ws;
  size_t off = 0;
  auto carve = [&](size_t bytes) -> void* {
    void* p = w + off; off += (bytes + 255) & ~(size_t)255; return p;
  };
  u16* hr_bf   = (u16*)carve((size_t)NB*NR*HID*2);
  u16* hv_bf   = (u16*)carve((size_t)NB*NV*HID*2);
  u16* hr2_bf  = (u16*)carve((size_t)NB*NR*HID*2);
  u16* uh_bf   = (u16*)carve((size_t)NB*NU*HID*2);
  u16* z_bf    = (u16*)carve((size_t)NB*512*2);
  u16* z1_bf   = (u16*)carve((size_t)NB*256*2);
  u16* wApk    = (u16*)carve(HID*HID*2);
  u16* wPpk    = (u16*)carve(HID*HID*2);
  u16* wCpk    = (u16*)carve(HID*HID*2);
  u16* woutC   = (u16*)carve(HID*HID*2);
  u16* fus1C   = (u16*)carve(256*HID*2);
  u16* fus2C   = (u16*)carve(HID*HID*2);
  u16* uav2C   = (u16*)carve(HID*HID*2);
  u16* fin1C   = (u16*)carve(512*256*2);
  u16* fin2C   = (u16*)carve(256*HID*2);
  float* wfs_a = (float*)carve(NH*HID*4);  float* wft_a = (float*)carve(NH*HID*4);
  float* wfs_p = (float*)carve(NH*HID*4);  float* wft_p = (float*)carve(NH*HID*4);
  float* wfs_c = (float*)carve(NH*HID*4);  float* wft_c = (float*)carve(NH*HID*4);
  float* c_a   = (float*)carve(NH*8*4);    float* d_a   = (float*)carve(NH*4);
  float* c_p   = (float*)carve(NH*8*4);    float* d_p   = (float*)carve(NH*4);
  float* c_c   = (float*)carve(NH*8*4);    float* d_c   = (float*)carve(NH*4);
  float* es_a  = (float*)carve((size_t)NB*NH*NR*4);
  float* et_a  = (float*)carve((size_t)NB*NH*NV*4);
  float* es_p  = (float*)carve((size_t)NB*NH*NR*4);
  float* et_p  = (float*)carve((size_t)NB*NH*NR*4);
  float* es_c  = (float*)carve((size_t)NB*NH*NR*4);
  float* et_c  = (float*)carve((size_t)NB*NH*NR*4);
  u16* whsTA   = (u16*)carve((size_t)HID*NB*NR*2);   // [H*32, B*NR] transposed
  u16* whsTP   = (u16*)carve((size_t)HID*NB*NR*2);
  u16* whsTC   = (u16*)carve((size_t)HID*NB*NR*2);
  u16* att     = (u16*)carve((size_t)NB*NH*NV*NR*2); // reused for all 3 GATs
  u16* g_bf    = (u16*)carve((size_t)NB*NV*HID*2);
  float* vrepr = (float*)carve((size_t)NB*NV*HID*4);
  u16* fused   = (u16*)carve((size_t)NB*NR*256*2);
  u16* fush    = (u16*)carve((size_t)NB*NR*HID*2);
  float* hfus  = (float*)carve((size_t)NB*NR*HID*4);
  float* uout  = (float*)carve((size_t)NB*NU*HID*4);

  // ---- prep: folds + weight packing (linear weights: plain cast, Bt = W) ----
  gat_prep<<<1, 512, 0, stream>>>(gaW, gaA, gaeW, gaeB, wfs_a, wft_a, c_a, d_a);
  gat_prep<<<1, 512, 0, stream>>>(gpW, gpA, gpeW, gpeB, wfs_p, wft_p, c_p, d_p);
  gat_prep<<<1, 512, 0, stream>>>(gcW, gcA, gceW, gceB, wfs_c, wft_c, c_c, d_c);
  pack_gatW<<<64, 256, 0, stream>>>(gaW, wApk);
  pack_gatW<<<64, 256, 0, stream>>>(gpW, wPpk);
  pack_gatW<<<64, 256, 0, stream>>>(gcW, wCpk);
  pack_cast<<<64, 256, 0, stream>>>(voW,  woutC, 128*128);
  pack_cast<<<128,256, 0, stream>>>(f1W,  fus1C, 128*256);
  pack_cast<<<64, 256, 0, stream>>>(f2W,  fus2C, 128*128);
  pack_cast<<<64, 256, 0, stream>>>(u2W,  uav2C, 128*128);
  pack_cast<<<512,256, 0, stream>>>(fn1W, fin1C, 256*512);
  pack_cast<<<128,256, 0, stream>>>(fn2W, fin2C, 128*256);

  // ---- input projections (relu(lin)) -> bf16 ----
  proj_small<<<(NB*NV*HID+255)/256, 256, 0, stream>>>(state,            5, NV, ST_STRIDE, vpW, vpB, hv_bf, 128, NB*NV*HID);
  proj_small<<<(NB*NR*HID+255)/256, 256, 0, stream>>>(state + NV*5,     5, NR, ST_STRIDE, rpW, rpB, hr_bf, 128, NB*NR*HID);
  proj_small<<<(NB*NR*HID+255)/256, 256, 0, stream>>>(state + NV*5,     5, NR, ST_STRIDE, r2W, r2B, hr2_bf,128, NB*NR*HID);
  proj_small<<<(NB*NU*HID+255)/256, 256, 0, stream>>>(state + (NV+NR)*5,5, NU, ST_STRIDE, u1W, u1B, uh_bf, 128, NB*NU*HID);
  proj_small<<<(NB*HID+255)/256,    256, 0, stream>>>(state + (NV+NR+NU)*5, 8, 1, ST_STRIDE, g1W, g1B, z_bf + 384, 512, NB*HID);

  // ---- per-node attention scores ----
  fold_score<<<(NB*NR*NH+255)/256, 256, 0, stream>>>(hr_bf,  wfs_a, es_a, NR, NB*NR*NH);
  fold_score<<<(NB*NV*NH+255)/256, 256, 0, stream>>>(hv_bf,  wft_a, et_a, NV, NB*NV*NH);
  fold_score<<<(NB*NR*NH+255)/256, 256, 0, stream>>>(hr2_bf, wfs_p, es_p, NR, NB*NR*NH);
  fold_score<<<(NB*NR*NH+255)/256, 256, 0, stream>>>(hr2_bf, wft_p, et_p, NR, NB*NR*NH);
  fold_score<<<(NB*NR*NH+255)/256, 256, 0, stream>>>(hr2_bf, wfs_c, es_c, NR, NB*NR*NH);
  fold_score<<<(NB*NR*NH+255)/256, 256, 0, stream>>>(hr2_bf, wft_c, et_c, NR, NB*NR*NH);

  auto gemm = [&](const u16* A, const u16* Bm, const float* bias, float* Cf, u16* Cb,
                  int M, int N, int K, int ldA, int ldB, int ldC, int innerN,
                  long aO, long aI, long bO, long bI, long cO, long cI,
                  int batches, int relu, int cbT) {
    dim3 grid((M + 127) / 128, N / 32, batches);
    gemm_bf16_wmma<<<grid, 128, 0, stream>>>(A, Bm, bias, Cf, Cb, M, N, K,
                                             ldA, ldB, ldC, innerN,
                                             aO, aI, bO, bI, cO, cI, relu, cbT);
  };

  // ---- Whs = h @ Wgat : stored TRANSPOSED [H*32, B*NR] for the hp GEMM ----
  gemm(hr_bf,  wApk, nullptr, nullptr, whsTA, NB*NR, 128, 128, 128, 128, NB*NR, 1, 0,0,0,0,0,0, 1, 0, 1);
  gemm(hr2_bf, wPpk, nullptr, nullptr, whsTP, NB*NR, 128, 128, 128, 128, NB*NR, 1, 0,0,0,0,0,0, 1, 0, 1);
  gemm(hr2_bf, wCpk, nullptr, nullptr, whsTC, NB*NR, 128, 128, 128, 128, NB*NR, 1, 0,0,0,0,0,0, 1, 0, 1);

  // ---- vehicle-route GAT (S=NR sources, T=NV targets) ----
  edge_score<<<dim3(NV/32, NR/32, NB), 256, 0, stream>>>(vr_edge, vr_mask, c_a, d_a,
                                                         es_a, et_a, att, NR, NV);
  softmax_rows<<<(NB*NH*NV)/8, 256, 0, stream>>>(att, es_a, et_a, NR, NV, 0, NB*NH*NV);
  // hp: batched 32 (b,h): att[bh][NV,NR] @ whsTA[h*32:, b*NR:]^T -> g[b][NV, h*32:]
  gemm(att, whsTA, nullptr, nullptr, g_bf, NV, 32, NR, NR, NB*NR, 128, NH,
       (long)NH*NV*NR, (long)NV*NR, (long)NR, 32L*NB*NR, (long)NV*128, 32, NB*NH, 0, 0);
  gemm(g_bf, woutC, voB, vrepr, nullptr, NB*NV, 128, 128, 128, 128, 128, 1, 0,0,0,0,0,0, 1, 0, 0);
  mean_rows<<<(NB*HID+255)/256, 256, 0, stream>>>(vrepr, z_bf, 0, NV);

  // ---- route-route physical GAT (att buffer reused) ----
  edge_score<<<dim3(NR/32, NR/32, NB), 256, 0, stream>>>(rr_edge, rr_mask, c_p, d_p,
                                                         es_p, et_p, att, NR, NR);
  softmax_rows<<<(NB*NH*NR)/8, 256, 0, stream>>>(att, es_p, et_p, NR, NR, 0, NB*NH*NR);
  gemm(att, whsTP, nullptr, nullptr, fused, NR, 32, NR, NR, NB*NR, 256, NH,
       (long)NH*NR*NR, (long)NR*NR, (long)NR, 32L*NB*NR, (long)NR*256, 32, NB*NH, 0, 0);

  // ---- route-route content GAT (no edge, no mask) ----
  softmax_rows<<<(NB*NH*NR)/8, 256, 0, stream>>>(att, es_c, et_c, NR, NR, 1, NB*NH*NR);
  gemm(att, whsTC, nullptr, nullptr, fused + 128, NR, 32, NR, NR, NB*NR, 256, NH,
       (long)NH*NR*NR, (long)NR*NR, (long)NR, 32L*NB*NR, (long)NR*256, 32, NB*NH, 0, 0);

  // ---- fusion MLP + pool ----
  gemm(fused, fus1C, f1B, nullptr, fush, NB*NR, 128, 256, 256, 256, 128, 1, 0,0,0,0,0,0, 1, 1, 0);
  gemm(fush,  fus2C, f2B, hfus, nullptr, NB*NR, 128, 128, 128, 128, 128, 1, 0,0,0,0,0,0, 1, 0, 0);
  mean_rows<<<(NB*HID+255)/256, 256, 0, stream>>>(hfus, z_bf, 128, NR);

  // ---- UAV path + pool ----
  gemm(uh_bf, uav2C, u2B, uout, nullptr, NB*NU, 128, 128, 128, 128, 128, 1, 0,0,0,0,0,0, 1, 0, 0);
  mean_rows<<<(NB*HID+255)/256, 256, 0, stream>>>(uout, z_bf, 256, NU);

  // ---- final MLP on z = [v_pool | r_pool | u_pool | g_enc] ----
  gemm(z_bf,  fin1C, fn1B, nullptr, z1_bf, NB, 256, 512, 512, 512, 256, 1, 0,0,0,0,0,0, 1, 1, 0);
  gemm(z1_bf, fin2C, fn2B, (float*)d_out, nullptr, NB, 128, 256, 256, 256, 128, 1, 0,0,0,0,0,0, 1, 0, 0);
}
